// BagleyDiT_8761733284138
// MI455X (gfx1250) — matrix operand order repaired and checked by reading, not compiled, not run on hardware
//
#include <hip/hip_runtime.h>
#include <hip/hip_bf16.h>
#include <stdint.h>
#include <stddef.h>

// Problem constants (must match reference)
#define B_  2
#define S_  1024
#define H_  1024
#define NH_ 16
#define HD_ 64
#define L_  256
#define C_  2048
#define E_  8
#define D_  2048
#define T_  (B_*S_)     // 2048 tokens
#define TL_ (B_*L_)     // 512 encoder tokens

typedef __attribute__((ext_vector_type(16))) __bf16 v16bf;
typedef __attribute__((ext_vector_type(8)))  float  v8f;

__device__ __forceinline__ uint16_t f2bf(float f) {
  uint32_t u = __float_as_uint(f);
  u += 0x7FFFu + ((u >> 16) & 1u);   // round to nearest even
  return (uint16_t)(u >> 16);
}
__device__ __forceinline__ float bf2f(uint16_t h) {
  return __uint_as_float(((uint32_t)h) << 16);
}
__device__ __forceinline__ uint32_t pack2bf(float lo, float hi) {
  return (uint32_t)f2bf(lo) | ((uint32_t)f2bf(hi) << 16);
}

// ---------------------------------------------------------------------------
// Generic tiled WMMA GEMM: C[M,N] = alpha * (A @ B) + bias, optional GELU.
// A: [M,K] row-major (bf16 or f32), lda.
// B: B_TRANS=0 -> source [K,N] (ldb), B_TRANS=1 -> source [N,K] (ldb).
// Requires M%128==0, N%64==0, K%32==0 (true for all call sites).
// Block = 256 threads = 8 waves, tile 128x64. Wave grid 4(M) x 2(N); each
// wave owns a 32x32 subtile: 2 A-frags x 2 B-frags -> 4 v_wmma_f32_16x16x32_bf16
// per K-step with full operand reuse. LDS: A row-major, B transposed to
// Bt[n][k]; all fragment gathers and LDS stores are aligned b32/b64 ops.
// Next K-tile is prefetched with WGP-scope global_prefetch_b8.
// ---------------------------------------------------------------------------
template<int A_F32, int B_F32, int B_TRANS, int OUT_F32, int GELU, int HAS_BIAS>
__global__ __launch_bounds__(256)
void gemm_wmma(const void* __restrict__ Av, int lda,
               const void* __restrict__ Bv, int ldb,
               const float* __restrict__ bias,
               void* __restrict__ Cv, int ldc,
               int Kdim, float alpha)
{
  __shared__ uint16_t As[128][36];  // row stride 36 elems (72B, 4B aligned)
  __shared__ uint16_t Bs[64][36];   // Bs[n][k]

  const int tid  = threadIdx.x;
  const int bm   = blockIdx.y * 128;
  const int bn   = blockIdx.x * 64;
  const int lane = tid & 31;
  const int wave = tid >> 5;
  const int wm   = (wave & 3) * 32;   // 4 waves along M (32 rows each)
  const int wn   = (wave >> 2) * 32;  // 2 waves along N (32 cols each)
  const int l15  = lane & 15;
  const int hi   = lane >> 4;

  v8f acc00 = {0.f,0.f,0.f,0.f,0.f,0.f,0.f,0.f};
  v8f acc01 = acc00, acc10 = acc00, acc11 = acc00;

  const int ar = tid >> 1;          // A stage: 128 rows, 2 threads/row
  const int ac = (tid & 1) * 16;    // 16 consecutive k each

  for (int k0 = 0; k0 < Kdim; k0 += 32) {
    // ---- stage A tile: 128 x 32 ----
    if (A_F32) {
      const float* Ap = (const float*)Av + (size_t)(bm + ar) * lda + (k0 + ac);
      #pragma unroll
      for (int i = 0; i < 16; i += 2)
        *(uint32_t*)&As[ar][ac + i] = pack2bf(Ap[i], Ap[i + 1]);
      if (k0 + 32 < Kdim) __builtin_prefetch(Ap + 32, 0, 3);
    } else {
      const uint32_t* Ap = (const uint32_t*)((const uint16_t*)Av +
                            (size_t)(bm + ar) * lda + (k0 + ac));
      #pragma unroll
      for (int i = 0; i < 8; ++i)
        *(uint32_t*)&As[ar][ac + 2 * i] = Ap[i];
      if (k0 + 32 < Kdim) __builtin_prefetch(Ap + 16, 0, 3);
    }
    // ---- stage B tile into Bs[n][k] ----
    if (B_TRANS) {                   // source is [N][K] (bf16 path used)
      const int r  = tid >> 2;       // n
      const int c0 = (tid & 3) * 8;  // k
      if (B_F32) {
        const float* Bp = (const float*)Bv + (size_t)(bn + r) * ldb + (k0 + c0);
        #pragma unroll
        for (int i = 0; i < 8; i += 2)
          *(uint32_t*)&Bs[r][c0 + i] = pack2bf(Bp[i], Bp[i + 1]);
        if (k0 + 32 < Kdim) __builtin_prefetch(Bp + 32, 0, 3);
      } else {
        const uint32_t* Bp = (const uint32_t*)((const uint16_t*)Bv +
                              (size_t)(bn + r) * ldb + (k0 + c0));
        #pragma unroll
        for (int i = 0; i < 4; ++i)
          *(uint32_t*)&Bs[r][c0 + 2 * i] = Bp[i];
        if (k0 + 32 < Kdim) __builtin_prefetch(Bp + 16, 0, 3);
      }
    } else {
      // source is [K][N]; transpose into LDS. Each thread handles a K-pair
      // (rows k0+2kp, k0+2kp+1) x 4 columns, packing the two K-values per
      // column into one aligned b32 LDS store: Bs[n][2kp] = {lo=k even, hi=k odd}.
      const int kp = tid >> 4;        // 0..15 k-pair
      const int n0 = (tid & 15) * 4;  // 4 n each
      if (B_F32) {
        const float* B0 = (const float*)Bv + (size_t)(k0 + 2 * kp) * ldb + (bn + n0);
        const float* B1 = B0 + ldb;
        #pragma unroll
        for (int i = 0; i < 4; ++i)
          *(uint32_t*)&Bs[n0 + i][2 * kp] = pack2bf(B0[i], B1[i]);
        if (k0 + 32 < Kdim) __builtin_prefetch(B0 + (size_t)32 * ldb, 0, 3);
      } else {
        const uint16_t* B0 = (const uint16_t*)Bv + (size_t)(k0 + 2 * kp) * ldb + (bn + n0);
        const uint16_t* B1 = B0 + ldb;
        #pragma unroll
        for (int i = 0; i < 4; ++i)
          *(uint32_t*)&Bs[n0 + i][2 * kp] = (uint32_t)B0[i] | ((uint32_t)B1[i] << 16);
        if (k0 + 32 < Kdim) __builtin_prefetch(B0 + (size_t)32 * ldb, 0, 3);
      }
    }
    __syncthreads();

    // ---- gather fragments (aligned b32 LDS reads), 4 WMMAs ----
    union Frag { uint32_t u[8]; v16bf v; } a0f, a1f, b0f, b1f;
    const uint32_t* Ar0 = (const uint32_t*)(&As[wm + l15][0]);
    const uint32_t* Ar1 = (const uint32_t*)(&As[wm + 16 + l15][0]);
    #pragma unroll
    for (int j = 0; j < 4; ++j) {
      a0f.u[j]     = Ar0[4 * hi + j];       // K = 8*hi + 2j,2j+1
      a0f.u[4 + j] = Ar0[8 + 4 * hi + j];   // K = 16 + 8*hi + 2j,2j+1
      a1f.u[j]     = Ar1[4 * hi + j];
      a1f.u[4 + j] = Ar1[8 + 4 * hi + j];
    }
    const uint32_t* Br0 = (const uint32_t*)(&Bs[wn + l15][0]);
    const uint32_t* Br1 = (const uint32_t*)(&Bs[wn + 16 + l15][0]);
    #pragma unroll
    for (int j = 0; j < 4; ++j) {
      b0f.u[j]     = Br0[4 * hi + j];       // K = 8*hi + 2j,2j+1
      b0f.u[4 + j] = Br0[8 + 4 * hi + j];   // K = 16 + 8*hi + 2j,2j+1
      b1f.u[j]     = Br1[4 * hi + j];
      b1f.u[4 + j] = Br1[8 + 4 * hi + j];
    }
#if defined(__HIP_DEVICE_COMPILE__)
    acc00 = __builtin_amdgcn_wmma_f32_16x16x32_bf16(false, a0f.v, false, b0f.v,
                                                    (short)0, acc00, false, false);
    acc01 = __builtin_amdgcn_wmma_f32_16x16x32_bf16(false, a0f.v, false, b1f.v,
                                                    (short)0, acc01, false, false);
    acc10 = __builtin_amdgcn_wmma_f32_16x16x32_bf16(false, a1f.v, false, b0f.v,
                                                    (short)0, acc10, false, false);
    acc11 = __builtin_amdgcn_wmma_f32_16x16x32_bf16(false, a1f.v, false, b1f.v,
                                                    (short)0, acc11, false, false);
#endif
    __syncthreads();
  }

  // ---- epilogue: C layout VGPR r -> row r + 8*hi, lane -> col l15 ----
  const int col0 = bn + wn + l15;
  const int col1 = col0 + 16;
  const float bias0 = HAS_BIAS ? bias[col0] : 0.f;
  const float bias1 = HAS_BIAS ? bias[col1] : 0.f;
  #pragma unroll
  for (int r = 0; r < 8; ++r) {
    const int row0 = bm + wm + 8 * hi + r;
    const int row1 = row0 + 16;
    float v00 = acc00[r] * alpha + bias0;
    float v01 = acc01[r] * alpha + bias1;
    float v10 = acc10[r] * alpha + bias0;
    float v11 = acc11[r] * alpha + bias1;
    if (GELU) {
      v00 = 0.5f * v00 * (1.f + erff(v00 * 0.70710678118654752f));
      v01 = 0.5f * v01 * (1.f + erff(v01 * 0.70710678118654752f));
      v10 = 0.5f * v10 * (1.f + erff(v10 * 0.70710678118654752f));
      v11 = 0.5f * v11 * (1.f + erff(v11 * 0.70710678118654752f));
    }
    if (OUT_F32) {
      float* Cp = (float*)Cv;
      Cp[(size_t)row0 * ldc + col0] = v00;
      Cp[(size_t)row0 * ldc + col1] = v01;
      Cp[(size_t)row1 * ldc + col0] = v10;
      Cp[(size_t)row1 * ldc + col1] = v11;
    } else {
      uint16_t* Cp = (uint16_t*)Cv;
      Cp[(size_t)row0 * ldc + col0] = f2bf(v00);
      Cp[(size_t)row0 * ldc + col1] = f2bf(v01);
      Cp[(size_t)row1 * ldc + col0] = f2bf(v10);
      Cp[(size_t)row1 * ldc + col1] = f2bf(v11);
    }
  }
}

// ---------------------------------------------------------------------------
// mod[b,j] = sum_i silu(cond[b,i]) * adaln_w[i,j] + adaln_b[j]
// ---------------------------------------------------------------------------
__global__ void mod_gemv_kernel(const float* __restrict__ cond,
                                const float* __restrict__ w,
                                const float* __restrict__ bias,
                                float* __restrict__ mod)
{
  int idx = blockIdx.x * 256 + threadIdx.x;      // B * 6H
  if (idx >= B_ * 6 * H_) return;
  int b = idx / (6 * H_);
  int j = idx - b * 6 * H_;
  float acc = 0.f;
  for (int i = 0; i < H_; ++i) {
    float c = cond[b * H_ + i];
    float s = c / (1.f + expf(-c));              // silu
    acc += s * w[(size_t)i * (6 * H_) + j];
  }
  mod[idx] = acc + bias[j];
}

// ---------------------------------------------------------------------------
// out_bf16[t,j] = ln(x[t,:])[j] * (add1 + scale[b*bstride+j]) + shift[b*bstride+j]
// ---------------------------------------------------------------------------
__global__ __launch_bounds__(256)
void ln_mod_kernel(const float* __restrict__ x, uint16_t* __restrict__ out,
                   const float* __restrict__ scale, const float* __restrict__ shift,
                   int bstride, float add1)
{
  __shared__ float red[256];
  const int t = blockIdx.x;
  const int b = t / S_;
  const float* xr = x + (size_t)t * H_;
  float lv[4];
  float s = 0.f;
  #pragma unroll
  for (int i = 0; i < 4; ++i) { lv[i] = xr[threadIdx.x + 256 * i]; s += lv[i]; }
  red[threadIdx.x] = s; __syncthreads();
  for (int o = 128; o > 0; o >>= 1) {
    if (threadIdx.x < o) red[threadIdx.x] += red[threadIdx.x + o];
    __syncthreads();
  }
  const float mean = red[0] / (float)H_;
  __syncthreads();
  float s2 = 0.f;
  #pragma unroll
  for (int i = 0; i < 4; ++i) { float d = lv[i] - mean; s2 += d * d; }
  red[threadIdx.x] = s2; __syncthreads();
  for (int o = 128; o > 0; o >>= 1) {
    if (threadIdx.x < o) red[threadIdx.x] += red[threadIdx.x + o];
    __syncthreads();
  }
  const float rstd = rsqrtf(red[0] / (float)H_ + 1e-5f);
  #pragma unroll
  for (int i = 0; i < 4; ++i) {
    int j = threadIdx.x + 256 * i;
    float sc = add1 + scale[(size_t)b * bstride + j];
    float sh = shift[(size_t)b * bstride + j];
    out[(size_t)t * H_ + j] = f2bf((lv[i] - mean) * rstd * sc + sh);
  }
}

// ---------------------------------------------------------------------------
// RoPE in place on q,k (bf16), layout [T, NH*HD]; one thread per (t,h,d<32).
// ---------------------------------------------------------------------------
__global__ void rope_kernel(uint16_t* __restrict__ q, uint16_t* __restrict__ k,
                            const float* __restrict__ cosb, const float* __restrict__ sinb)
{
  int idx = blockIdx.x * 256 + threadIdx.x;      // T*NH*32
  if (idx >= T_ * NH_ * 32) return;
  const int d = idx & 31;
  const int h = (idx >> 5) & (NH_ - 1);
  const int t = idx >> 9;
  const int s = t & (S_ - 1);
  const size_t base = (size_t)t * H_ + h * HD_;
  const float c1 = cosb[s * HD_ + d],      sn1 = sinb[s * HD_ + d];
  const float c2 = cosb[s * HD_ + d + 32], sn2 = sinb[s * HD_ + d + 32];
  {
    float x1 = bf2f(q[base + d]), x2 = bf2f(q[base + d + 32]);
    q[base + d]      = f2bf(x1 * c1 - x2 * sn1);
    q[base + d + 32] = f2bf(x2 * c2 + x1 * sn2);
  }
  {
    float x1 = bf2f(k[base + d]), x2 = bf2f(k[base + d + 32]);
    k[base + d]      = f2bf(x1 * c1 - x2 * sn1);
    k[base + d + 32] = f2bf(x2 * c2 + x1 * sn2);
  }
}

// ---------------------------------------------------------------------------
// Row softmax: s[row, 0:cols] (f32) -> p (bf16), same leading dim = cols.
// ---------------------------------------------------------------------------
__global__ __launch_bounds__(256)
void softmax_kernel(const float* __restrict__ s, uint16_t* __restrict__ p, int cols)
{
  __shared__ float red[256];
  const int row = blockIdx.x;
  const float* sr = s + (size_t)row * cols;
  float m = -1e30f;
  for (int c = threadIdx.x; c < cols; c += 256) m = fmaxf(m, sr[c]);
  red[threadIdx.x] = m; __syncthreads();
  for (int o = 128; o > 0; o >>= 1) {
    if (threadIdx.x < o) red[threadIdx.x] = fmaxf(red[threadIdx.x], red[threadIdx.x + o]);
    __syncthreads();
  }
  m = red[0]; __syncthreads();
  float sum = 0.f;
  for (int c = threadIdx.x; c < cols; c += 256) sum += expf(sr[c] - m);
  red[threadIdx.x] = sum; __syncthreads();
  for (int o = 128; o > 0; o >>= 1) {
    if (threadIdx.x < o) red[threadIdx.x] += red[threadIdx.x + o];
    __syncthreads();
  }
  const float inv = 1.f / red[0];
  for (int c = threadIdx.x; c < cols; c += 256)
    p[(size_t)row * cols + c] = f2bf(expf(sr[c] - m) * inv);
}

// ---------------------------------------------------------------------------
// out[t,n] = xin[t,n] + gate[b,n] * y[t,n]   (gate==nullptr -> 1)
// ---------------------------------------------------------------------------
__global__ void resgate_kernel(const float* __restrict__ xin, const float* __restrict__ y,
                               const float* __restrict__ gate, int gstride,
                               float* __restrict__ out)
{
  int idx = blockIdx.x * 256 + threadIdx.x;
  if (idx >= T_ * H_) return;
  const int n = idx & (H_ - 1);
  const int b = idx / (S_ * H_);
  const float g = gate ? gate[(size_t)b * gstride + n] : 1.f;
  out[idx] = xin[idx] + g * y[idx];
}

// acc[t,n] = beta*acc[t,n] + cw[t,e] * y[t,n]
__global__ void moe_axpy_kernel(const float* __restrict__ y, const float* __restrict__ cw,
                                int e, float beta, float* __restrict__ acc)
{
  int idx = blockIdx.x * 256 + threadIdx.x;
  if (idx >= T_ * H_) return;
  const int t = idx >> 10;   // / H_
  acc[idx] = beta * acc[idx] + cw[(size_t)t * E_ + e] * y[idx];
}

// ---------------------------------------------------------------------------
// Gating: logits = h @ gate_w, softmax(8), top-2 renormalized combine weights.
// ---------------------------------------------------------------------------
__global__ void gate_kernel(const uint16_t* __restrict__ h, const float* __restrict__ gw,
                            float* __restrict__ probs, float* __restrict__ cw)
{
  __shared__ float part[64];
  const int t = blockIdx.x;
  const int e = threadIdx.x & 7;
  const int chunk = threadIdx.x >> 3;   // 0..7, 128 elems each
  float acc = 0.f;
  for (int i = chunk * 128; i < chunk * 128 + 128; ++i)
    acc += bf2f(h[(size_t)t * H_ + i]) * gw[(size_t)i * E_ + e];
  part[threadIdx.x] = acc;
  __syncthreads();
  if (threadIdx.x < 8) {
    float lg = 0.f;
    for (int c = 0; c < 8; ++c) lg += part[c * 8 + threadIdx.x];
    part[threadIdx.x] = lg;
  }
  __syncthreads();
  if (threadIdx.x == 0) {
    float m = part[0];
    for (int i = 1; i < E_; ++i) m = fmaxf(m, part[i]);
    float p[E_]; float sum = 0.f;
    for (int i = 0; i < E_; ++i) { p[i] = expf(part[i] - m); sum += p[i]; }
    for (int i = 0; i < E_; ++i) { p[i] /= sum; probs[(size_t)t * E_ + i] = p[i]; }
    int i1 = 0;
    for (int i = 1; i < E_; ++i) if (p[i] > p[i1]) i1 = i;
    int i2 = -1;
    for (int i = 0; i < E_; ++i) if (i != i1 && (i2 < 0 || p[i] > p[i2])) i2 = i;
    const float ws = p[i1] + p[i2];
    for (int i = 0; i < E_; ++i) cw[(size_t)t * E_ + i] = 0.f;
    cw[(size_t)t * E_ + i1] = p[i1] / ws;
    cw[(size_t)t * E_ + i2] = p[i2] / ws;
  }
}

// ---------------------------------------------------------------------------
// aux = E * sum_e (count_e / B) * (probsum_e / T)   (deterministic reduction)
// ---------------------------------------------------------------------------
__global__ __launch_bounds__(256)
void aux_kernel(const float* __restrict__ probs, const float* __restrict__ cw,
                float* __restrict__ out)
{
  __shared__ float sp[256];
  __shared__ float sc[256];
  float lp[E_], lc[E_];
  #pragma unroll
  for (int e = 0; e < E_; ++e) { lp[e] = 0.f; lc[e] = 0.f; }
  for (int t = threadIdx.x; t < T_; t += 256) {
    #pragma unroll
    for (int e = 0; e < E_; ++e) {
      lp[e] += probs[(size_t)t * E_ + e];
      lc[e] += (cw[(size_t)t * E_ + e] > 0.f) ? 1.f : 0.f;
    }
  }
  float aux = 0.f;
  for (int e = 0; e < E_; ++e) {
    __syncthreads();
    sp[threadIdx.x] = lp[e];
    sc[threadIdx.x] = lc[e];
    __syncthreads();
    for (int o = 128; o > 0; o >>= 1) {
      if (threadIdx.x < o) { sp[threadIdx.x] += sp[threadIdx.x + o]; sc[threadIdx.x] += sc[threadIdx.x + o]; }
      __syncthreads();
    }
    if (threadIdx.x == 0) aux += (sc[0] / (float)B_) * (sp[0] / (float)T_);
  }
  if (threadIdx.x == 0) out[0] = (float)E_ * aux;
}

// ---------------------------------------------------------------------------
extern "C" void kernel_launch(void* const* d_in, const int* in_sizes, int n_in,
                              void* d_out, int out_size, void* d_ws, size_t ws_size,
                              hipStream_t stream)
{
  (void)in_sizes; (void)n_in; (void)out_size; (void)ws_size;
  const float* x    = (const float*)d_in[0];
  const float* cond = (const float*)d_in[1];
  const float* enc  = (const float*)d_in[2];
  const float* rcos = (const float*)d_in[3];
  const float* rsin = (const float*)d_in[4];
  const float* adw  = (const float*)d_in[5];
  const float* adb  = (const float*)d_in[6];
  const float* qw   = (const float*)d_in[7];  const float* qb  = (const float*)d_in[8];
  const float* kw   = (const float*)d_in[9];  const float* kb  = (const float*)d_in[10];
  const float* vw   = (const float*)d_in[11]; const float* vb  = (const float*)d_in[12];
  const float* ow   = (const float*)d_in[13]; const float* ob  = (const float*)d_in[14];
  const float* cqw  = (const float*)d_in[15]; const float* cqb = (const float*)d_in[16];
  const float* ckw  = (const float*)d_in[17]; const float* ckb = (const float*)d_in[18];
  const float* cvw  = (const float*)d_in[19]; const float* cvb = (const float*)d_in[20];
  const float* cow  = (const float*)d_in[21]; const float* cob = (const float*)d_in[22];
  const float* cng  = (const float*)d_in[23]; const float* cnb = (const float*)d_in[24];
  const float* gw   = (const float*)d_in[25];
  const float* ew1  = (const float*)d_in[26]; const float* eb1 = (const float*)d_in[27];
  const float* ew2  = (const float*)d_in[28]; const float* eb2 = (const float*)d_in[29];

  float* xout = (float*)d_out;                 // running x, then aux at [T*H]

  // Workspace layout (~48 MB)
  char* ws = (char*)d_ws;
  size_t o = 0;
  auto alloc = [&](size_t bytes) -> void* {
    void* p = (void*)(ws + o);
    o += (bytes + 255) & ~(size_t)255;
    return p;
  };
  float*    modb = (float*)   alloc((size_t)B_ * 6 * H_ * 4);
  uint16_t* hbuf = (uint16_t*)alloc((size_t)T_ * H_ * 2);
  uint16_t* qbuf = (uint16_t*)alloc((size_t)T_ * H_ * 2);
  uint16_t* kbuf = (uint16_t*)alloc((size_t)T_ * H_ * 2);
  uint16_t* vbuf = (uint16_t*)alloc((size_t)T_ * H_ * 2);
  float*    sbuf = (float*)   alloc((size_t)S_ * S_ * 4);
  uint16_t* pbuf = (uint16_t*)alloc((size_t)S_ * S_ * 2);
  float*    ybuf = (float*)   alloc((size_t)T_ * H_ * 4);
  float*    tbuf = (float*)   alloc((size_t)T_ * H_ * 4);
  uint16_t* actb = (uint16_t*)alloc((size_t)T_ * D_ * 2);
  uint16_t* lkb  = (uint16_t*)alloc((size_t)TL_ * H_ * 2);
  uint16_t* lvb  = (uint16_t*)alloc((size_t)TL_ * H_ * 2);
  float*    prb  = (float*)   alloc((size_t)T_ * E_ * 4);
  float*    cwb  = (float*)   alloc((size_t)T_ * E_ * 4);

  const dim3 blk(256);

  // 1) AdaLN modulation vector
  mod_gemv_kernel<<<(B_ * 6 * H_ + 255) / 256, blk, 0, stream>>>(cond, adw, adb, modb);

  // 2) h1 = ln(x)*(1+sc1)+sh1     (mod row: [sh1|sc1|g1|sh2|sc2|g2])
  ln_mod_kernel<<<T_, blk, 0, stream>>>(x, hbuf, modb + H_, modb, 6 * H_, 1.f);

  // 3) QKV projections (bf16 out)
  gemm_wmma<0,1,0,0,0,1><<<dim3(H_/64, T_/128), blk, 0, stream>>>(hbuf, H_, qw, H_, qb, qbuf, H_, H_, 1.f);
  gemm_wmma<0,1,0,0,0,1><<<dim3(H_/64, T_/128), blk, 0, stream>>>(hbuf, H_, kw, H_, kb, kbuf, H_, H_, 1.f);
  gemm_wmma<0,1,0,0,0,1><<<dim3(H_/64, T_/128), blk, 0, stream>>>(hbuf, H_, vw, H_, vb, vbuf, H_, H_, 1.f);

  // 4) RoPE on q,k
  rope_kernel<<<(T_ * NH_ * 32) / 256, blk, 0, stream>>>(qbuf, kbuf, rcos, rsin);

  // 5) self-attention per (b, head)
  for (int b = 0; b < B_; ++b)
    for (int h = 0; h < NH_; ++h) {
      const size_t hoff = (size_t)b * S_ * H_ + (size_t)h * HD_;
      gemm_wmma<0,0,1,1,0,0><<<dim3(S_/64, S_/128), blk, 0, stream>>>(
          qbuf + hoff, H_, kbuf + hoff, H_, nullptr, sbuf, S_, HD_, 0.125f);
      softmax_kernel<<<S_, blk, 0, stream>>>(sbuf, pbuf, S_);
      gemm_wmma<0,0,0,1,0,0><<<dim3(HD_/64, S_/128), blk, 0, stream>>>(
          pbuf, S_, vbuf + hoff, H_, nullptr, ybuf + hoff, H_, S_, 1.f);
    }

  // 6) x = x + g1 * (a @ ow + ob)
  gemm_wmma<1,1,0,1,0,1><<<dim3(H_/64, T_/128), blk, 0, stream>>>(ybuf, H_, ow, H_, ob, tbuf, H_, H_, 1.f);
  resgate_kernel<<<(T_ * H_) / 256, blk, 0, stream>>>(x, tbuf, modb + 2 * H_, 6 * H_, xout);

  // 7) cross-attention: hc = ln(x)*cn_g + cn_b
  ln_mod_kernel<<<T_, blk, 0, stream>>>(xout, hbuf, cng, cnb, 0, 0.f);
  gemm_wmma<0,1,0,0,0,1><<<dim3(H_/64, T_/128),  blk, 0, stream>>>(hbuf, H_, cqw, H_, cqb, qbuf, H_, H_, 1.f);
  gemm_wmma<1,1,0,0,0,1><<<dim3(H_/64, TL_/128), blk, 0, stream>>>(enc, C_, ckw, H_, ckb, lkb, H_, C_, 1.f);
  gemm_wmma<1,1,0,0,0,1><<<dim3(H_/64, TL_/128), blk, 0, stream>>>(enc, C_, cvw, H_, cvb, lvb, H_, C_, 1.f);
  for (int b = 0; b < B_; ++b)
    for (int h = 0; h < NH_; ++h) {
      const size_t qoff = (size_t)b * S_ * H_ + (size_t)h * HD_;
      const size_t koff = (size_t)b * L_ * H_ + (size_t)h * HD_;
      gemm_wmma<0,0,1,1,0,0><<<dim3(L_/64, S_/128), blk, 0, stream>>>(
          qbuf + qoff, H_, lkb + koff, H_, nullptr, sbuf, L_, HD_, 0.125f);
      softmax_kernel<<<S_, blk, 0, stream>>>(sbuf, pbuf, L_);
      gemm_wmma<0,0,0,1,0,0><<<dim3(HD_/64, S_/128), blk, 0, stream>>>(
          pbuf, L_, lvb + koff, H_, nullptr, ybuf + qoff, H_, L_, 1.f);
    }
  gemm_wmma<1,1,0,1,0,1><<<dim3(H_/64, T_/128), blk, 0, stream>>>(ybuf, H_, cow, H_, cob, tbuf, H_, H_, 1.f);
  resgate_kernel<<<(T_ * H_) / 256, blk, 0, stream>>>(xout, tbuf, nullptr, 0, xout);

  // 8) MoE: h2 = ln(x)*(1+sc2)+sh2, gate, aux, dense per-expert FFN + combine
  ln_mod_kernel<<<T_, blk, 0, stream>>>(xout, hbuf, modb + 4 * H_, modb + 3 * H_, 6 * H_, 1.f);
  gate_kernel<<<T_, dim3(64), 0, stream>>>(hbuf, gw, prb, cwb);
  aux_kernel<<<1, blk, 0, stream>>>(prb, cwb, xout + (size_t)T_ * H_);
  for (int e = 0; e < E_; ++e) {
    gemm_wmma<0,1,0,0,1,1><<<dim3(D_/64, T_/128), blk, 0, stream>>>(
        hbuf, H_, ew1 + (size_t)e * H_ * D_, D_, eb1 + (size_t)e * D_, actb, D_, H_, 1.f);
    gemm_wmma<0,1,0,1,0,1><<<dim3(H_/64, T_/128), blk, 0, stream>>>(
        actb, D_, ew2 + (size_t)e * D_ * H_, H_, eb2 + (size_t)e * H_, ybuf, H_, D_, 1.f);
    moe_axpy_kernel<<<(T_ * H_) / 256, blk, 0, stream>>>(ybuf, cwb, e, (e == 0) ? 0.f : 1.f, tbuf);
  }
  resgate_kernel<<<(T_ * H_) / 256, blk, 0, stream>>>(xout, tbuf, modb + 5 * H_, 6 * H_, xout);
}
